// Block1_64080912056607
// MI455X (gfx1250) — matrix operand order repaired and checked by reading, not compiled, not run on hardware
//
#include <hip/hip_runtime.h>
#include <math.h>

typedef __attribute__((ext_vector_type(16))) _Float16 v16h;
typedef __attribute__((ext_vector_type(8)))  float    v8f;

#define NWS 7
#define NH 4
#define HD 32
#define CDIM 128
#define NN 49
#define NP 64
#define DFF 512

// ---------------- WMMA fragment helpers (wave32, gfx1250 layouts) ----------------

__device__ __forceinline__ v8f wmma_f16(v16h a, v16h b, v8f c) {
  return __builtin_amdgcn_wmma_f32_16x16x32_f16(false, a, false, b, (short)0, c, false, false);
}

// A matrix 16x32 f16, row-major source [16 x >=32], lda halfs.
__device__ __forceinline__ v16h frag_a(const _Float16* p, int lda) {
  int lane = threadIdx.x & 31;
  int m = lane & 15;
  int kh = (lane >> 4) << 3;
  union { v16h v; unsigned u[8]; } f;
#pragma unroll
  for (int i = 0; i < 8; ++i) {
    int k = ((i >> 2) << 4) + kh + ((i & 3) << 1);
    f.u[i] = *(const unsigned*)(p + m * lda + k);
  }
  return f.v;
}

// B matrix 32x16 f16 from transposed (N-major) storage Bt[N][K], ldk halfs.
__device__ __forceinline__ v16h frag_bt(const _Float16* p, int ldk) {
  int lane = threadIdx.x & 31;
  int n = lane & 15;
  int kh = (lane >> 4) << 3;
  union { v16h v; unsigned u[8]; } f;
#pragma unroll
  for (int i = 0; i < 8; ++i) {
    f.u[i] = *(const unsigned*)(p + n * ldk + ((i >> 2) << 4) + kh + ((i & 3) << 1));
  }
  return f.v;
}

__device__ __forceinline__ int reg3(int p) { return p < 49 ? 0 : (p < 53 ? 1 : 2); }

// ---------------- Kernel 0: weight transpose+convert f32[K][N] -> f16[N][K] ----------------
__global__ void k_wconv(const float* __restrict__ src, _Float16* __restrict__ dst, int K, int Nn) {
  int i = blockIdx.x * blockDim.x + threadIdx.x;
  if (i < K * Nn) {
    int n = i / K, k = i - n * K;
    dst[n * K + k] = (_Float16)src[k * Nn + n];
  }
}

// ---------------- Kernel 1: LN1 + window partition -> f16 [2048][64][128] ----------------
__global__ void k_ln1_part(const float* __restrict__ x, const float* __restrict__ lw,
                           const float* __restrict__ lb, _Float16* __restrict__ h) {
  int wave = threadIdx.x >> 5;
  int lane = threadIdx.x & 31;
  int row = blockIdx.x * 8 + wave;            // over (win, n) : 2048*64
  int win = row >> 6;
  int n = row & 63;
  _Float16* dst = h + (win * NP + n) * CDIM;
  if (n >= NN) {
    *(unsigned long long*)(dst + lane * 4) = 0ULL;
    return;
  }
  int b = win >> 6, wi = win & 63, wy = wi >> 3, wx = wi & 7;
  int r = n / NWS, c = n - r * NWS;
  const float* src = x + ((b * 3136) + (wy * NWS + r) * 56 + (wx * NWS + c)) * CDIM;
  float4 xv = *(const float4*)(src + lane * 4);
  float s = xv.x + xv.y + xv.z + xv.w;
  float s2 = xv.x * xv.x + xv.y * xv.y + xv.z * xv.z + xv.w * xv.w;
#pragma unroll
  for (int off = 1; off < 32; off <<= 1) {
    s  += __shfl_xor(s,  off, 32);
    s2 += __shfl_xor(s2, off, 32);
  }
  float mu = s * (1.0f / 128.0f);
  float var = s2 * (1.0f / 128.0f) - mu * mu;
  float inv = rsqrtf(var + 1e-5f);
  float vx[4] = {xv.x, xv.y, xv.z, xv.w};
  union { _Float16 hv[4]; unsigned long long u; } o;
#pragma unroll
  for (int j = 0; j < 4; ++j) {
    int ch = lane * 4 + j;
    o.hv[j] = (_Float16)((vx[j] - mu) * inv * lw[ch] + lb[ch]);
  }
  *(unsigned long long*)(dst + lane * 4) = o.u;
}

// ---------------- Kernel 2: qkv GEMM per window (uniform q/k/v phases) ----------------
__global__ void k_qkv(const _Float16* __restrict__ h, const _Float16* __restrict__ wT,
                      const float* __restrict__ bias, _Float16* __restrict__ q,
                      _Float16* __restrict__ k, _Float16* __restrict__ v) {
  int win = blockIdx.x;
  int wv = threadIdx.x >> 5;
  int lane = threadIdx.x & 31;
  int ml = lane & 15;
  const _Float16* A = h + win * NP * CDIM + wv * 16 * CDIM;
  v16h a0 = frag_a(A, CDIM), a1 = frag_a(A + 32, CDIM), a2 = frag_a(A + 64, CDIM), a3 = frag_a(A + 96, CDIM);
  int mbase = wv * 16 + ((lane >> 4) << 3);

  // phase 0: Q  (cols 0..127)
  for (int nt = 0; nt < 8; ++nt) {
    v8f acc = {};
    const _Float16* B = wT + nt * 16 * CDIM;
    acc = wmma_f16(a0, frag_bt(B, CDIM), acc);
    acc = wmma_f16(a1, frag_bt(B + 32, CDIM), acc);
    acc = wmma_f16(a2, frag_bt(B + 64, CDIM), acc);
    acc = wmma_f16(a3, frag_bt(B + 96, CDIM), acc);
    int hh = nt >> 1;
    int d = ((nt & 1) << 4) + ml;
    float bv = bias[nt * 16 + ml];
    int base = ((win * NH + hh) * NP + mbase) * HD + d;
#pragma unroll
    for (int r2 = 0; r2 < 8; ++r2)
      q[base + r2 * HD] = (_Float16)((acc[r2] + bv) * 0.1767766953f);
  }
  // phase 1: K  (cols 128..255)
  for (int nt = 0; nt < 8; ++nt) {
    v8f acc = {};
    const _Float16* B = wT + (8 + nt) * 16 * CDIM;
    acc = wmma_f16(a0, frag_bt(B, CDIM), acc);
    acc = wmma_f16(a1, frag_bt(B + 32, CDIM), acc);
    acc = wmma_f16(a2, frag_bt(B + 64, CDIM), acc);
    acc = wmma_f16(a3, frag_bt(B + 96, CDIM), acc);
    int hh = nt >> 1;
    int d = ((nt & 1) << 4) + ml;
    float bv = bias[128 + nt * 16 + ml];
    int base = ((win * NH + hh) * NP + mbase) * HD + d;
#pragma unroll
    for (int r2 = 0; r2 < 8; ++r2)
      k[base + r2 * HD] = (_Float16)(mbase + r2 < NN ? acc[r2] + bv : 0.0f);
  }
  // phase 2: V  (cols 256..383), stored [h][d][m]: 8 consecutive m per lane -> b128 store
  for (int nt = 0; nt < 8; ++nt) {
    v8f acc = {};
    const _Float16* B = wT + (16 + nt) * 16 * CDIM;
    acc = wmma_f16(a0, frag_bt(B, CDIM), acc);
    acc = wmma_f16(a1, frag_bt(B + 32, CDIM), acc);
    acc = wmma_f16(a2, frag_bt(B + 64, CDIM), acc);
    acc = wmma_f16(a3, frag_bt(B + 96, CDIM), acc);
    int hh = nt >> 1;
    int d = ((nt & 1) << 4) + ml;
    float bv = bias[256 + nt * 16 + ml];
    int base = ((win * NH + hh) * HD + d) * NP + mbase;
    union { _Float16 hv[8]; uint4 u; } pk;
#pragma unroll
    for (int r2 = 0; r2 < 8; ++r2)
      pk.hv[r2] = (_Float16)(mbase + r2 < NN ? acc[r2] + bv : 0.0f);
    *(uint4*)(v + base) = pk.u;
  }
}

// ---------------- Kernel 3: attention per (window, head) ----------------
__global__ void k_attn(const _Float16* __restrict__ q, const _Float16* __restrict__ k,
                       const _Float16* __restrict__ v, const float* __restrict__ rpb,
                       _Float16* __restrict__ o) {
  __shared__ _Float16 Pl[NP * NP];
  __shared__ float brpb[169 * NH];
  int win = blockIdx.x >> 2;
  int hh = blockIdx.x & 3;
  for (int i = threadIdx.x; i < 169 * NH; i += blockDim.x) brpb[i] = rpb[i];
  __syncthreads();

  int wv = threadIdx.x >> 5;
  int lane = threadIdx.x & 31;
  const _Float16* qb = q + (win * NH + hh) * NP * HD;
  const _Float16* kb = k + (win * NH + hh) * NP * HD;
  const _Float16* vb = v + (win * NH + hh) * HD * NP;

  v16h aq = frag_a(qb + wv * 16 * HD, HD);
  v8f s[4];
#pragma unroll
  for (int j = 0; j < 4; ++j) {
    v8f z = {};
    s[j] = wmma_f16(aq, frag_bt(kb + j * 16 * HD, HD), z);
  }

  int wi = win & 63, wy = wi >> 3, wx = wi & 7;
  int nb = wv * 16 + ((lane >> 4) << 3);
  int ml = lane & 15;
#pragma unroll
  for (int j = 0; j < 4; ++j) {
    int m = j * 16 + ml;
    if (m >= NN) {
#pragma unroll
      for (int r2 = 0; r2 < 8; ++r2) s[j][r2] = -3.0e38f;
    } else {
      int rm = m / NWS, cm = m - rm * NWS;
      int regm = reg3(wy * NWS + rm) * 3 + reg3(wx * NWS + cm);
#pragma unroll
      for (int r2 = 0; r2 < 8; ++r2) {
        int n = nb + r2;
        if (n < NN) {
          int rn = n / NWS, cn = n - rn * NWS;
          float bv = brpb[((rn - rm + 6) * 13 + (cn - cm + 6)) * NH + hh];
          int regn = reg3(wy * NWS + rn) * 3 + reg3(wx * NWS + cn);
          s[j][r2] += bv + ((regn != regm) ? -100.0f : 0.0f);
        }
      }
    }
  }

  // in-register softmax: each row lives in 16 lanes (one half-wave) at fixed VGPR index
#pragma unroll
  for (int r2 = 0; r2 < 8; ++r2) {
    float mx = fmaxf(fmaxf(s[0][r2], s[1][r2]), fmaxf(s[2][r2], s[3][r2]));
#pragma unroll
    for (int off = 1; off < 16; off <<= 1) mx = fmaxf(mx, __shfl_xor(mx, off, 32));
    float e0 = __expf(s[0][r2] - mx);
    float e1 = __expf(s[1][r2] - mx);
    float e2 = __expf(s[2][r2] - mx);
    float e3 = __expf(s[3][r2] - mx);
    float sum = e0 + e1 + e2 + e3;
#pragma unroll
    for (int off = 1; off < 16; off <<= 1) sum += __shfl_xor(sum, off, 32);
    float inv = 1.0f / sum;
    _Float16* pr = &Pl[(nb + r2) * NP + ml];
    pr[0]  = (_Float16)(e0 * inv);
    pr[16] = (_Float16)(e1 * inv);
    pr[32] = (_Float16)(e2 * inv);
    pr[48] = (_Float16)(e3 * inv);
  }
  __syncthreads();

  // O = P @ V   (A = P from LDS, B = V^T from global)
#pragma unroll
  for (int dt = 0; dt < 2; ++dt) {
    v8f acc = {};
    acc = wmma_f16(frag_a(&Pl[wv * 16 * NP], NP),      frag_bt(vb + dt * 16 * NP, NP), acc);
    acc = wmma_f16(frag_a(&Pl[wv * 16 * NP + 32], NP), frag_bt(vb + dt * 16 * NP + 32, NP), acc);
    int base = (win * NP + nb) * CDIM + hh * HD + dt * 16 + ml;
#pragma unroll
    for (int r2 = 0; r2 < 8; ++r2) {
      float val = (nb + r2 < NN) ? acc[r2] : 0.0f;
      o[base + r2 * CDIM] = (_Float16)val;
    }
  }
}

// ---------------- Kernel 4: proj + window reverse + residual -> f32 x_attn ----------------
__global__ void k_proj(const _Float16* __restrict__ o, const _Float16* __restrict__ pT,
                       const float* __restrict__ pb, const float* __restrict__ x,
                       float* __restrict__ xo) {
  int win = blockIdx.x;
  int wv = threadIdx.x >> 5;
  int lane = threadIdx.x & 31;
  const _Float16* A = o + win * NP * CDIM + wv * 16 * CDIM;
  v16h a0 = frag_a(A, CDIM), a1 = frag_a(A + 32, CDIM), a2 = frag_a(A + 64, CDIM), a3 = frag_a(A + 96, CDIM);
  int b = win >> 6, wi = win & 63, wy = wi >> 3, wx = wi & 7;
  int nb = wv * 16 + ((lane >> 4) << 3);
  for (int nt = 0; nt < 8; ++nt) {
    v8f acc = {};
    const _Float16* B = pT + nt * 16 * CDIM;
    acc = wmma_f16(a0, frag_bt(B, CDIM), acc);
    acc = wmma_f16(a1, frag_bt(B + 32, CDIM), acc);
    acc = wmma_f16(a2, frag_bt(B + 64, CDIM), acc);
    acc = wmma_f16(a3, frag_bt(B + 96, CDIM), acc);
    int ch = nt * 16 + (lane & 15);
    float bv = pb[ch];
#pragma unroll
    for (int r2 = 0; r2 < 8; ++r2) {
      int n = nb + r2;
      if (n < NN) {
        int r = n / NWS, c = n - r * NWS;
        int idx = ((b * 3136) + (wy * NWS + r) * 56 + (wx * NWS + c)) * CDIM + ch;
        xo[idx] = acc[r2] + bv + x[idx];
      }
    }
  }
}

// ---------------- Kernel 5: LN2 + fc1 + exact GELU -> f16 m1 [rows][512] ----------------
__global__ void k_mlp1(const float* __restrict__ xo, const float* __restrict__ lw,
                       const float* __restrict__ lb, const _Float16* __restrict__ w1T,
                       const float* __restrict__ b1, _Float16* __restrict__ m1) {
  __shared__ _Float16 Al[64 * CDIM];
  int wv = threadIdx.x >> 5;
  int lane = threadIdx.x & 31;
  int rowbase = blockIdx.x * 64;
  for (int rr = wv; rr < 64; rr += 4) {
    const float* src = xo + (rowbase + rr) * CDIM;
    float4 xv = *(const float4*)(src + lane * 4);
    float s = xv.x + xv.y + xv.z + xv.w;
    float s2 = xv.x * xv.x + xv.y * xv.y + xv.z * xv.z + xv.w * xv.w;
#pragma unroll
    for (int off = 1; off < 32; off <<= 1) {
      s  += __shfl_xor(s,  off, 32);
      s2 += __shfl_xor(s2, off, 32);
    }
    float mu = s * (1.0f / 128.0f);
    float inv = rsqrtf(s2 * (1.0f / 128.0f) - mu * mu + 1e-5f);
    float vx[4] = {xv.x, xv.y, xv.z, xv.w};
#pragma unroll
    for (int j = 0; j < 4; ++j) {
      int ch = lane * 4 + j;
      Al[rr * CDIM + ch] = (_Float16)((vx[j] - mu) * inv * lw[ch] + lb[ch]);
    }
  }
  __syncthreads();
  const _Float16* A = &Al[wv * 16 * CDIM];
  v16h a0 = frag_a(A, CDIM), a1 = frag_a(A + 32, CDIM), a2 = frag_a(A + 64, CDIM), a3 = frag_a(A + 96, CDIM);
  int nb = wv * 16 + ((lane >> 4) << 3);
  for (int nt = 0; nt < 32; ++nt) {
    v8f acc = {};
    const _Float16* B = w1T + nt * 16 * CDIM;
    acc = wmma_f16(a0, frag_bt(B, CDIM), acc);
    acc = wmma_f16(a1, frag_bt(B + 32, CDIM), acc);
    acc = wmma_f16(a2, frag_bt(B + 64, CDIM), acc);
    acc = wmma_f16(a3, frag_bt(B + 96, CDIM), acc);
    int ch = nt * 16 + (lane & 15);
    float bv = b1[ch];
    int base = (rowbase + nb) * DFF + ch;
#pragma unroll
    for (int r2 = 0; r2 < 8; ++r2) {
      float val = acc[r2] + bv;
      float g = 0.5f * val * (1.0f + erff(val * 0.70710678118f));  // exact GELU
      m1[base + r2 * DFF] = (_Float16)g;
    }
  }
}

// ---------------- Kernel 6: fc2 + residual -> f32 out ----------------
__global__ void k_mlp2(const _Float16* __restrict__ m1, const _Float16* __restrict__ w2T,
                       const float* __restrict__ b2, const float* __restrict__ xo,
                       float* __restrict__ out) {
  int wv = threadIdx.x >> 5;
  int lane = threadIdx.x & 31;
  int rowbase = blockIdx.x * 64;
  v8f acc[8];
#pragma unroll
  for (int nt = 0; nt < 8; ++nt) { v8f z = {}; acc[nt] = z; }
  const _Float16* A = m1 + (rowbase + wv * 16) * DFF;
  for (int kt = 0; kt < 16; ++kt) {
    v16h a = frag_a(A + kt * 32, DFF);
#pragma unroll
    for (int nt = 0; nt < 8; ++nt) {
      acc[nt] = wmma_f16(a, frag_bt(w2T + nt * 16 * DFF + kt * 32, DFF), acc[nt]);
    }
  }
  int nb = wv * 16 + ((lane >> 4) << 3);
#pragma unroll
  for (int nt = 0; nt < 8; ++nt) {
    int ch = nt * 16 + (lane & 15);
    float bv = b2[ch];
    int base = (rowbase + nb) * CDIM + ch;
#pragma unroll
    for (int r2 = 0; r2 < 8; ++r2) {
      int idx = base + r2 * CDIM;
      out[idx] = acc[nt][r2] + bv + xo[idx];
    }
  }
}

// ---------------- Host launcher ----------------
extern "C" void kernel_launch(void* const* d_in, const int* in_sizes, int n_in,
                              void* d_out, int out_size, void* d_ws, size_t ws_size,
                              hipStream_t stream) {
  const float* x      = (const float*)d_in[0];
  const float* qkv_w  = (const float*)d_in[1];
  const float* qkv_b  = (const float*)d_in[2];
  const float* proj_w = (const float*)d_in[3];
  const float* proj_b = (const float*)d_in[4];
  const float* rpb    = (const float*)d_in[5];
  const float* ln1_w  = (const float*)d_in[6];
  const float* ln1_b  = (const float*)d_in[7];
  const float* ln2_w  = (const float*)d_in[8];
  const float* ln2_b  = (const float*)d_in[9];
  const float* mlp_w1 = (const float*)d_in[10];
  const float* mlp_b1 = (const float*)d_in[11];
  const float* mlp_w2 = (const float*)d_in[12];
  const float* mlp_b2 = (const float*)d_in[13];
  float* out = (float*)d_out;
  char* ws = (char*)d_ws;

  // workspace layout (bytes)
  _Float16* qkvT  = (_Float16*)(ws + 0);                 //  98304  : [384][128]
  _Float16* projT = (_Float16*)(ws + 98304);             //  32768  : [128][128]
  _Float16* w1T   = (_Float16*)(ws + 131072);            // 131072  : [512][128]
  _Float16* w2T   = (_Float16*)(ws + 262144);            // 131072  : [128][512]
  _Float16* hbuf  = (_Float16*)(ws + 393216);            // 33.5MB  : h (K1,K2) then o (K3,K4)
  _Float16* qb    = (_Float16*)(ws + 33947648);          // 3x33.5MB: q,k,v ; later aliased as m1
  _Float16* kb    = qb + 2048 * NH * NP * HD;
  _Float16* vb    = kb + 2048 * NH * NP * HD;
  _Float16* m1    = qb;                                  // alias (q/k/v dead after K3)
  float*    xattn = (float*)(ws + 136708096);            // 51.4MB f32

  k_wconv<<<(128 * 384 + 255) / 256, 256, 0, stream>>>(qkv_w, qkvT, 128, 384);
  k_wconv<<<(128 * 128 + 255) / 256, 256, 0, stream>>>(proj_w, projT, 128, 128);
  k_wconv<<<(128 * 512 + 255) / 256, 256, 0, stream>>>(mlp_w1, w1T, 128, 512);
  k_wconv<<<(512 * 128 + 255) / 256, 256, 0, stream>>>(mlp_w2, w2T, 512, 128);

  k_ln1_part<<<16384, 256, 0, stream>>>(x, ln1_w, ln1_b, hbuf);
  k_qkv<<<2048, 128, 0, stream>>>(hbuf, qkvT, qkv_b, qb, kb, vb);
  k_attn<<<8192, 128, 0, stream>>>(qb, kb, vb, rpb, hbuf);
  k_proj<<<2048, 128, 0, stream>>>(hbuf, projT, proj_b, x, xattn);
  k_mlp1<<<1568, 128, 0, stream>>>(xattn, ln2_w, ln2_b, w1T, mlp_b1, m1);
  k_mlp2<<<1568, 128, 0, stream>>>(m1, w2T, mlp_b2, xattn, out);

  (void)in_sizes; (void)n_in; (void)out_size; (void)ws_size;
}
